// SwinTransformerBlock_SingleTask_19430432047581
// MI455X (gfx1250) — matrix operand order repaired and checked by reading, not compiled, not run on hardware
//
#include <hip/hip_runtime.h>

typedef __attribute__((ext_vector_type(16))) _Float16 v16h;
typedef __attribute__((ext_vector_type(8)))  _Float16 v8h;
typedef __attribute__((ext_vector_type(8)))  float    v8f;
typedef int v4i __attribute__((vector_size(4 * sizeof(int))));

#define WMMA_F16(a, b, c) \
  __builtin_amdgcn_wmma_f32_16x16x32_f16(false, (a), false, (b), (short)0, (c), false, false)

// ---------------- async global->LDS (CDNA5) with fallback ----------------
#if defined(__has_builtin)
#if __has_builtin(__builtin_amdgcn_global_load_async_to_lds_b128) && \
    __has_builtin(__builtin_amdgcn_s_wait_asynccnt)
#define HAVE_ASYNC_LDS 1
#endif
#endif
#ifndef HAVE_ASYNC_LDS
#define HAVE_ASYNC_LDS 0
#endif

__device__ __forceinline__ void async_copy16(const _Float16* g, _Float16* l) {
#if HAVE_ASYNC_LDS
  __builtin_amdgcn_global_load_async_to_lds_b128(
      (__attribute__((address_space(1))) v4i*)(g),
      (__attribute__((address_space(3))) v4i*)(l), 0, 0);
#else
  *(v8h*)l = *(const v8h*)g;
#endif
}

template <int N>
__device__ __forceinline__ void async_wait_c() {
#if HAVE_ASYNC_LDS
  __builtin_amdgcn_s_wait_asynccnt(N);
#endif
}

// ---------------- constants ----------------
#define BQ   8
#define HH   64
#define WW   64
#define DIM  512
#define WS   8
#define SHIFTV 4
#define NHEAD 16
#define HD   32
#define NTOK 64            // tokens per window
#define NWIN 512           // B * nW
#define MROWS 32768        // total tokens
#define NEXP 4
#define HID  2048
#define GH   256

// ---------------- small utility kernels ----------------
__global__ __launch_bounds__(256) void cvt_f32_to_f16(const float* __restrict__ in,
                                                      _Float16* __restrict__ out, int n) {
  int i = blockIdx.x * 256 + threadIdx.x;
  if (i < n) out[i] = (_Float16)in[i];
}

__global__ __launch_bounds__(256) void build_qkv_bias(const float* __restrict__ qb,
                                                      const float* __restrict__ vb,
                                                      float* __restrict__ out) {
  int i = blockIdx.x * 256 + threadIdx.x;
  if (i < 3 * DIM) out[i] = (i < DIM) ? qb[i] : ((i < 2 * DIM) ? 0.0f : vb[i - 2 * DIM]);
}

// cyclic shift (-4,-4) + window partition + f32 -> f16
__global__ __launch_bounds__(256) void win_partition(const float* __restrict__ x,
                                                     _Float16* __restrict__ xw) {
  int row = blockIdx.x;                 // window-token row 0..32767
  int win = row >> 6, n = row & 63;
  int b = win >> 6, widx = win & 63;
  int wy = widx >> 3, wx = widx & 7;
  int sh = wy * WS + (n >> 3), sw = wx * WS + (n & 7);   // rolled-frame coords
  int oh = (sh + SHIFTV) & (HH - 1), ow = (sw + SHIFTV) & (WW - 1);
  const float* src = x + ((size_t)b * (HH * WW) + oh * WW + ow) * DIM;
  _Float16* dst = xw + (size_t)row * DIM;
  for (int c = threadIdx.x; c < DIM; c += 256) dst[c] = (_Float16)src[c];
}

// continuous relative position bias MLP; store 16*sigmoid(o) directly.
__device__ __forceinline__ float logcoord(float r) {
  float t = r * (8.0f / 7.0f);
  float s = (t > 0.0f) ? 1.0f : ((t < 0.0f) ? -1.0f : 0.0f);
  return s * log2f(fabsf(t) + 1.0f) * 0.3333333333f;   // /log2(8)
}

__global__ __launch_bounds__(256) void cpb_kernel(const float* __restrict__ w1,
                                                  const float* __restrict__ b1,
                                                  const float* __restrict__ w2,
                                                  float* __restrict__ tbl16) {
  __shared__ float hbuf[512];
  int i = blockIdx.x;                                  // 0..224
  float c0 = logcoord((float)(i / 15) - 7.0f);
  float c1 = logcoord((float)(i % 15) - 7.0f);
  for (int j = threadIdx.x; j < 512; j += 256)
    hbuf[j] = fmaxf(0.0f, c0 * w1[j * 2 + 0] + c1 * w1[j * 2 + 1] + b1[j]);
  __syncthreads();
  if (threadIdx.x < NHEAD) {
    int h = threadIdx.x;
    float o = 0.0f;
    for (int j = 0; j < 512; ++j) o += hbuf[j] * w2[h * 512 + j];
    tbl16[i * NHEAD + h] = 16.0f / (1.0f + __expf(-o));
  }
}

// ---------------- generic WMMA GEMM: D = A[M,K] * Wt[N,K]^T + bias ----------------
// 256 threads = 8 waves. Block tile 128x64, K-step 32, double-buffered async LDS.
// Wave w -> rows [w*16, w*16+16) x all 64 cols (4 WMMA accumulators, A-frag reuse).
#define EPI_F32_BIAS 0
#define EPI_F16_BIAS 1
#define EPI_F16_RELU 2
#define EPI_F16_GELU 3
#define EPI_MOE      4

template <int EPI>
__global__ __launch_bounds__(256) void gemm_wmma(const _Float16* __restrict__ A,
                                                 const _Float16* __restrict__ Wt,
                                                 const float* __restrict__ bias,
                                                 void* __restrict__ Dout,
                                                 int M, int Nout, int K,
                                                 const float* __restrict__ wts,
                                                 int expert, int accum) {
  __shared__ __align__(16) _Float16 Ab[2][128][40];   // [row][k], 80B rows (16B-aligned)
  __shared__ __align__(16) _Float16 Bb[2][64][40];    // [col][k] (K-contiguous)
  const int tid  = threadIdx.x;
  const int lane = tid & 31;
  const int wave = tid >> 5;            // 16-row strip 0..7
  const int bm = blockIdx.x * 128;
  const int bn = blockIdx.y * 64;
  const int fr = lane & 15, fs = lane >> 4;

  v8f acc[4];
#pragma unroll
  for (int tn = 0; tn < 4; ++tn) acc[tn] = v8f{};

  // tile chunk mapping: 16B per chunk, 256 chunks per 64x32 tile
  const int cr = tid >> 2, ckc = (tid & 3) * 8;
  const int KT = K >> 5;

  auto issue = [&](int t, int buf) {
    const int k0 = t << 5;
    async_copy16(A + (size_t)(bm + cr) * K + k0 + ckc,       &Ab[buf][cr][ckc]);
    async_copy16(A + (size_t)(bm + 64 + cr) * K + k0 + ckc,  &Ab[buf][64 + cr][ckc]);
    async_copy16(Wt + (size_t)(bn + cr) * K + k0 + ckc,      &Bb[buf][cr][ckc]);
  };

  issue(0, 0);
  for (int t = 0; t < KT; ++t) {
    const int cur = t & 1;
    __syncthreads();                       // prev compute done; buf cur^1 free
    if (t + 1 < KT) {
      issue(t + 1, cur ^ 1);
      async_wait_c<3>();                   // tile t complete, tile t+1 in flight
    } else {
      async_wait_c<0>();
    }
    __syncthreads();                       // all waves' tile t visible

    v16h a;
#pragma unroll
    for (int j = 0; j < 8; ++j) {
      a[j]     = Ab[cur][wave * 16 + fr][fs * 16 + j];
      a[j + 8] = Ab[cur][wave * 16 + fr][fs * 16 + 8 + j];
    }
#pragma unroll
    for (int tn = 0; tn < 4; ++tn) {
      v16h b;
#pragma unroll
      for (int j = 0; j < 16; ++j) b[j] = Bb[cur][tn * 16 + fr][fs * 16 + j];
      acc[tn] = WMMA_F16(a, b, acc[tn]);
    }
  }

#pragma unroll
  for (int tn = 0; tn < 4; ++tn) {
    const int col = bn + tn * 16 + fr;
    const float bb = bias[col];
#pragma unroll
    for (int r = 0; r < 8; ++r) {
      const int row = bm + wave * 16 + fs * 8 + r;
      float xv = acc[tn][r] + bb;
      size_t idx = (size_t)row * Nout + col;
      if constexpr (EPI == EPI_F32_BIAS) {
        ((float*)Dout)[idx] = xv;
      } else if constexpr (EPI == EPI_F16_BIAS) {
        ((_Float16*)Dout)[idx] = (_Float16)xv;
      } else if constexpr (EPI == EPI_F16_RELU) {
        ((_Float16*)Dout)[idx] = (_Float16)fmaxf(xv, 0.0f);
      } else if constexpr (EPI == EPI_F16_GELU) {
        ((_Float16*)Dout)[idx] = (_Float16)(0.5f * xv * (1.0f + erff(xv * 0.70710678f)));
      } else {  // EPI_MOE: D (f32) (+)= wts[row,expert] * (acc + bias)
        float* D = (float*)Dout;
        float rv = wts[(size_t)row * NEXP + expert] * xv;
        if (accum) rv += D[idx];
        D[idx] = rv;
      }
    }
  }
}

// ---------------- attention: one (window, head) per 2-wave block ----------------
__device__ __forceinline__ int regionof(int g) { return g < 56 ? 0 : (g < 60 ? 1 : 2); }

__global__ __launch_bounds__(64) void attn_kernel(const _Float16* __restrict__ qkv,
                                                  const float* __restrict__ logit_scale,
                                                  const float* __restrict__ tbl16,
                                                  _Float16* __restrict__ attnout) {
  __shared__ _Float16 qn[64][40];
  __shared__ _Float16 kn[64][40];
  __shared__ _Float16 vs[64][40];
  __shared__ float    S[64][65];
  __shared__ _Float16 P[64][72];
  const int blk = blockIdx.x;
  const int win = blk >> 4;
  const int h   = blk & 15;
  const int tid = threadIdx.x;
  const int wave = tid >> 5, lane = tid & 31;
  const int fr = lane & 15, fs = lane >> 4;

  {  // load + cosine-normalize q,k; copy v
    const _Float16* base = qkv + (size_t)(win * NTOK + tid) * (3 * DIM) + h * HD;
    float tmp[HD]; float ss = 0.0f;
#pragma unroll
    for (int d = 0; d < HD; ++d) { float f = (float)base[d]; tmp[d] = f; ss += f * f; }
    float inv = 1.0f / fmaxf(sqrtf(ss), 1e-12f);
#pragma unroll
    for (int d = 0; d < HD; ++d) qn[tid][d] = (_Float16)(tmp[d] * inv);
    ss = 0.0f;
#pragma unroll
    for (int d = 0; d < HD; ++d) { float f = (float)base[DIM + d]; tmp[d] = f; ss += f * f; }
    inv = 1.0f / fmaxf(sqrtf(ss), 1e-12f);
#pragma unroll
    for (int d = 0; d < HD; ++d) kn[tid][d] = (_Float16)(tmp[d] * inv);
#pragma unroll
    for (int d = 0; d < HD; ++d) vs[tid][d] = base[2 * DIM + d];
  }
  __syncthreads();

  const float scale = __expf(fminf(logit_scale[h], 4.60517019f));
  const int widx = win & 63, wy = widx >> 3, wx = widx & 7;

  // S = qn (64x32) @ kn^T (32x64), fused scale + CRPB + shift mask
  for (int tm = 0; tm < 4; ++tm) {
    v16h a;
#pragma unroll
    for (int j = 0; j < 8; ++j) {
      a[j]     = qn[tm * 16 + fr][fs * 16 + j];
      a[j + 8] = qn[tm * 16 + fr][fs * 16 + 8 + j];
    }
    for (int tn2 = 0; tn2 < 2; ++tn2) {
      int tn = wave * 2 + tn2;
      v16h b;
#pragma unroll
      for (int j = 0; j < 16; ++j) b[j] = kn[tn * 16 + fr][fs * 16 + j];
      v8f c = {};
      c = WMMA_F16(a, b, c);
      int colm = tn * 16 + fr;
      int r1 = colm >> 3, c1 = colm & 7;
      int regm = regionof(wy * 8 + r1) * 3 + regionof(wx * 8 + c1);
#pragma unroll
      for (int r = 0; r < 8; ++r) {
        int rown = tm * 16 + fs * 8 + r;
        int r0 = rown >> 3, c0 = rown & 7;
        int idx = (r0 - r1 + 7) * 15 + (c0 - c1 + 7);
        float bv = tbl16[idx * NHEAD + h];
        int regn = regionof(wy * 8 + r0) * 3 + regionof(wx * 8 + c0);
        float mk = (regn != regm) ? -100.0f : 0.0f;
        S[rown][colm] = c[r] * scale + bv + mk;
      }
    }
  }
  __syncthreads();

  {  // per-row softmax -> P (f16)
    int n = tid;
    float mx = -1e30f;
    for (int m = 0; m < 64; ++m) mx = fmaxf(mx, S[n][m]);
    float sum = 0.0f;
    for (int m = 0; m < 64; ++m) { float e = __expf(S[n][m] - mx); sum += e; S[n][m] = e; }
    float inv = 1.0f / sum;
    for (int m = 0; m < 64; ++m) P[n][m] = (_Float16)(S[n][m] * inv);
  }
  __syncthreads();

  // out = P (64x64) @ v (64x32); wave handles 16 of 32 head dims
  for (int tm = 0; tm < 4; ++tm) {
    v8f c = {};
    for (int k0 = 0; k0 < 2; ++k0) {
      v16h a, b;
#pragma unroll
      for (int j = 0; j < 8; ++j) {
        a[j]     = P[tm * 16 + fr][k0 * 32 + fs * 16 + j];
        a[j + 8] = P[tm * 16 + fr][k0 * 32 + fs * 16 + 8 + j];
      }
#pragma unroll
      for (int j = 0; j < 16; ++j) b[j] = vs[k0 * 32 + fs * 16 + j][wave * 16 + fr];
      c = WMMA_F16(a, b, c);
    }
#pragma unroll
    for (int r = 0; r < 8; ++r) {
      int rown = tm * 16 + fs * 8 + r;
      attnout[(size_t)(win * NTOK + rown) * DIM + h * HD + wave * 16 + fr] = (_Float16)c[r];
    }
  }
}

// ---------------- window reverse + roll + postnorm residual ----------------
__global__ __launch_bounds__(256) void postattn_ln(const float* __restrict__ projout,
                                                   const float* __restrict__ x,
                                                   const float* __restrict__ nw,
                                                   const float* __restrict__ nb,
                                                   float* __restrict__ x1f,
                                                   _Float16* __restrict__ x1h) {
  __shared__ float red[256];
  int row = blockIdx.x;
  int b = row >> 12, pos = row & 4095;
  int oh = pos >> 6, ow = pos & 63;
  int sh = (oh - SHIFTV) & (HH - 1), sw = (ow - SHIFTV) & (WW - 1);
  int win = b * 64 + (sh >> 3) * 8 + (sw >> 3);
  int n = (sh & 7) * 8 + (sw & 7);
  const float* src = projout + ((size_t)win * NTOK + n) * DIM;
  const float* xs  = x + (size_t)row * DIM;
  int t = threadIdx.x;
  float s = 0.0f;
  for (int c = t; c < DIM; c += 256) s += src[c];
  red[t] = s; __syncthreads();
  for (int st = 128; st > 0; st >>= 1) { if (t < st) red[t] += red[t + st]; __syncthreads(); }
  float mean = red[0] / (float)DIM; __syncthreads();
  float v = 0.0f;
  for (int c = t; c < DIM; c += 256) { float d = src[c] - mean; v += d * d; }
  red[t] = v; __syncthreads();
  for (int st = 128; st > 0; st >>= 1) { if (t < st) red[t] += red[t + st]; __syncthreads(); }
  float inv = rsqrtf(red[0] / (float)DIM + 1e-5f);
  for (int c = t; c < DIM; c += 256) {
    float val = xs[c] + (src[c] - mean) * inv * nw[c] + nb[c];
    x1f[(size_t)row * DIM + c] = val;
    x1h[(size_t)row * DIM + c] = (_Float16)val;
  }
}

// ---------------- gate finish: softmax(256), gm=g*aw, l2, softmax(4) ----------------
__global__ __launch_bounds__(256) void gate_finish(const _Float16* __restrict__ g,
                                                   const float* __restrict__ falog,
                                                   const float* __restrict__ l2w,
                                                   const float* __restrict__ l2b,
                                                   float* __restrict__ wts) {
  __shared__ float red[256];
  __shared__ float gm[256];
  __shared__ float logits[NEXP];
  int row = blockIdx.x;
  int t = threadIdx.x;
  float fl = falog[(size_t)row * GH + t];
  red[t] = fl; __syncthreads();
  for (int st = 128; st > 0; st >>= 1) { if (t < st) red[t] = fmaxf(red[t], red[t + st]); __syncthreads(); }
  float mx = red[0]; __syncthreads();
  float e = __expf(fl - mx);
  red[t] = e; __syncthreads();
  for (int st = 128; st > 0; st >>= 1) { if (t < st) red[t] += red[t + st]; __syncthreads(); }
  float aw = e / red[0]; __syncthreads();
  gm[t] = (float)g[(size_t)row * GH + t] * aw;
  __syncthreads();
  for (int ee = 0; ee < NEXP; ++ee) {
    red[t] = gm[t] * l2w[ee * GH + t]; __syncthreads();
    for (int st = 128; st > 0; st >>= 1) { if (t < st) red[t] += red[t + st]; __syncthreads(); }
    if (t == 0) logits[ee] = red[0] + l2b[ee];
    __syncthreads();
  }
  if (t == 0) {
    float m = fmaxf(fmaxf(logits[0], logits[1]), fmaxf(logits[2], logits[3]));
    float s = 0.0f, ex[NEXP];
    for (int ee = 0; ee < NEXP; ++ee) { ex[ee] = __expf(logits[ee] - m); s += ex[ee]; }
    for (int ee = 0; ee < NEXP; ++ee) wts[(size_t)row * NEXP + ee] = ex[ee] / s;
  }
}

// ---------------- final: out = x1 + LN(moe) ----------------
__global__ __launch_bounds__(256) void final_ln(const float* __restrict__ moe,
                                                const float* __restrict__ x1f,
                                                const float* __restrict__ nw,
                                                const float* __restrict__ nb,
                                                float* __restrict__ out) {
  __shared__ float red[256];
  int row = blockIdx.x;
  const float* src = moe + (size_t)row * DIM;
  int t = threadIdx.x;
  float s = 0.0f;
  for (int c = t; c < DIM; c += 256) s += src[c];
  red[t] = s; __syncthreads();
  for (int st = 128; st > 0; st >>= 1) { if (t < st) red[t] += red[t + st]; __syncthreads(); }
  float mean = red[0] / (float)DIM; __syncthreads();
  float v = 0.0f;
  for (int c = t; c < DIM; c += 256) { float d = src[c] - mean; v += d * d; }
  red[t] = v; __syncthreads();
  for (int st = 128; st > 0; st >>= 1) { if (t < st) red[t] += red[t + st]; __syncthreads(); }
  float inv = rsqrtf(red[0] / (float)DIM + 1e-5f);
  for (int c = t; c < DIM; c += 256)
    out[(size_t)row * DIM + c] = x1f[(size_t)row * DIM + c] + (src[c] - mean) * inv * nw[c] + nb[c];
}

// ---------------- host side ----------------
extern "C" void kernel_launch(void* const* d_in, const int* in_sizes, int n_in,
                              void* d_out, int out_size, void* d_ws, size_t ws_size,
                              hipStream_t stream) {
  const float* x        = (const float*)d_in[0];
  const float* qkv_w    = (const float*)d_in[1];
  const float* q_bias   = (const float*)d_in[2];
  const float* v_bias   = (const float*)d_in[3];
  const float* lscale   = (const float*)d_in[4];
  const float* cpb_w1   = (const float*)d_in[5];
  const float* cpb_b1   = (const float*)d_in[6];
  const float* cpb_w2   = (const float*)d_in[7];
  const float* proj_w   = (const float*)d_in[8];
  const float* proj_b   = (const float*)d_in[9];
  const float* norm1_w  = (const float*)d_in[10];
  const float* norm1_b  = (const float*)d_in[11];
  const float* norm2_w  = (const float*)d_in[12];
  const float* norm2_b  = (const float*)d_in[13];
  const float* exp_w1   = (const float*)d_in[14];
  const float* exp_b1   = (const float*)d_in[15];
  const float* exp_w2   = (const float*)d_in[16];
  const float* exp_b2   = (const float*)d_in[17];
  const float* l1_w     = (const float*)d_in[18];
  const float* l1_b     = (const float*)d_in[19];
  const float* fa_w     = (const float*)d_in[20];
  const float* fa_b     = (const float*)d_in[21];
  const float* l2_w     = (const float*)d_in[22];
  const float* l2_b     = (const float*)d_in[23];
  const float* task_w   = (const float*)d_in[24];
  const float* task_b   = (const float*)d_in[25];
  (void)in_sizes; (void)n_in; (void)out_size; (void)ws_size;

  char* ws = (char*)d_ws;
  // persistent region
  const size_t OFF_QKVW = 0;                       // 1536*512 f16
  const size_t OFF_PROJW = OFF_QKVW + 1572864;     // 512*512 f16
  const size_t OFF_TASKW = OFF_PROJW + 524288;
  const size_t OFF_L1W   = OFF_TASKW + 524288;     // 256*512 f16
  const size_t OFF_FAW   = OFF_L1W + 262144;       // 256*256 f16
  const size_t OFF_EW1   = OFF_FAW + 131072;       // 4*2048*512 f16
  const size_t OFF_EW2   = OFF_EW1 + 8388608;      // 4*512*2048 f16
  const size_t OFF_QKVB  = OFF_EW2 + 8388608;      // 1536 f32
  const size_t OFF_TBL   = OFF_QKVB + 8192;        // 225*16 f32
  const size_t OFF_WTS   = OFF_TBL + 16384;        // 32768*4 f32
  // big regions (with aliasing)
  const size_t OFF_A   = 32ull << 20;              // xw / attnout f16 (33.5MB)
  const size_t OFF_B   = OFF_A + 33554432;         // qkv f16 (100.7MB); later t/g/fa; later h
  const size_t OFF_C   = OFF_B + 100663296;        // projout f32 (67.1MB); h spills into here
  const size_t OFF_X1F = OFF_C + 67108864;         // x1 f32
  const size_t OFF_X1H = OFF_X1F + 67108864;       // x1 f16
  const size_t OFF_MOE = OFF_X1H + 33554432;       // moe f32
  const size_t OFF_T  = OFF_B;
  const size_t OFF_G  = OFF_B + 33554432;
  const size_t OFF_FA = OFF_G + 16777216;
  const size_t OFF_H  = OFF_B;

  _Float16* qkvw_h = (_Float16*)(ws + OFF_QKVW);
  _Float16* projw_h = (_Float16*)(ws + OFF_PROJW);
  _Float16* taskw_h = (_Float16*)(ws + OFF_TASKW);
  _Float16* l1w_h  = (_Float16*)(ws + OFF_L1W);
  _Float16* faw_h  = (_Float16*)(ws + OFF_FAW);
  _Float16* ew1_h  = (_Float16*)(ws + OFF_EW1);
  _Float16* ew2_h  = (_Float16*)(ws + OFF_EW2);
  float* qkv_bias  = (float*)(ws + OFF_QKVB);
  float* tbl16     = (float*)(ws + OFF_TBL);
  float* wts       = (float*)(ws + OFF_WTS);
  _Float16* xw     = (_Float16*)(ws + OFF_A);
  _Float16* attno  = (_Float16*)(ws + OFF_A);
  _Float16* qkv    = (_Float16*)(ws + OFF_B);
  float* projout   = (float*)(ws + OFF_C);
  float* x1f       = (float*)(ws + OFF_X1F);
  _Float16* x1h    = (_Float16*)(ws + OFF_X1H);
  float* moe       = (float*)(ws + OFF_MOE);
  _Float16* t_h    = (_Float16*)(ws + OFF_T);
  _Float16* g_h    = (_Float16*)(ws + OFF_G);
  float* fa_f      = (float*)(ws + OFF_FA);
  _Float16* h_h    = (_Float16*)(ws + OFF_H);

  auto cvt = [&](const float* in, _Float16* out, int n) {
    cvt_f32_to_f16<<<(n + 255) / 256, 256, 0, stream>>>(in, out, n);
  };
  // weight down-conversion
  cvt(qkv_w, qkvw_h, 1536 * 512);
  cvt(proj_w, projw_h, 512 * 512);
  cvt(task_w, taskw_h, 512 * 512);
  cvt(l1_w, l1w_h, GH * 512);
  cvt(fa_w, faw_h, GH * GH);
  cvt(exp_w1, ew1_h, NEXP * HID * DIM);
  cvt(exp_w2, ew2_h, NEXP * DIM * HID);
  build_qkv_bias<<<6, 256, 0, stream>>>(q_bias, v_bias, qkv_bias);
  cpb_kernel<<<225, 256, 0, stream>>>(cpb_w1, cpb_b1, cpb_w2, tbl16);
  win_partition<<<MROWS, 256, 0, stream>>>(x, xw);

  // qkv projection
  gemm_wmma<EPI_F16_BIAS><<<dim3(MROWS / 128, 1536 / 64), 256, 0, stream>>>(
      xw, qkvw_h, qkv_bias, (void*)qkv, MROWS, 1536, DIM, nullptr, 0, 0);
  // attention
  attn_kernel<<<NWIN * NHEAD, 64, 0, stream>>>(qkv, lscale, tbl16, attno);
  // output projection
  gemm_wmma<EPI_F32_BIAS><<<dim3(MROWS / 128, DIM / 64), 256, 0, stream>>>(
      attno, projw_h, proj_b, (void*)projout, MROWS, DIM, DIM, nullptr, 0, 0);
  // window reverse + LN + residual
  postattn_ln<<<MROWS, 256, 0, stream>>>(projout, x, norm1_w, norm1_b, x1f, x1h);

  // gating path
  gemm_wmma<EPI_F16_BIAS><<<dim3(MROWS / 128, DIM / 64), 256, 0, stream>>>(
      x1h, taskw_h, task_b, (void*)t_h, MROWS, DIM, DIM, nullptr, 0, 0);
  gemm_wmma<EPI_F16_RELU><<<dim3(MROWS / 128, GH / 64), 256, 0, stream>>>(
      t_h, l1w_h, l1_b, (void*)g_h, MROWS, GH, DIM, nullptr, 0, 0);
  gemm_wmma<EPI_F32_BIAS><<<dim3(MROWS / 128, GH / 64), 256, 0, stream>>>(
      g_h, faw_h, fa_b, (void*)fa_f, MROWS, GH, GH, nullptr, 0, 0);
  gate_finish<<<MROWS, 256, 0, stream>>>(g_h, fa_f, l2_w, l2_b, wts);

  // experts (h aliases qkv/t/g/fa region, which is dead now)
  for (int e = 0; e < NEXP; ++e) {
    gemm_wmma<EPI_F16_GELU><<<dim3(MROWS / 128, HID / 64), 256, 0, stream>>>(
        x1h, ew1_h + (size_t)e * HID * DIM, exp_b1 + (size_t)e * HID,
        (void*)h_h, MROWS, HID, DIM, nullptr, 0, 0);
    gemm_wmma<EPI_MOE><<<dim3(MROWS / 128, DIM / 64), 256, 0, stream>>>(
        h_h, ew2_h + (size_t)e * DIM * HID, exp_b2 + (size_t)e * DIM,
        (void*)moe, MROWS, DIM, HID, wts, e, e > 0 ? 1 : 0);
  }
  // out = x1 + LN(moe)
  final_ln<<<MROWS, 256, 0, stream>>>(moe, x1f, norm2_w, norm2_b, (float*)d_out);
}
